// FourierBasisTRF_27135603376406
// MI455X (gfx1250) — matrix operand order repaired and checked by reading, not compiled, not run on hardware
//
#include <hip/hip_runtime.h>
#include <math.h>

typedef __attribute__((ext_vector_type(16))) _Float16 v16h;
typedef __attribute__((ext_vector_type(8)))  float    v8f;

#define N_BASIS   129
#define MAX_N     64
#define K_DIM     128          // 2*MAX_N
#define M_TOTAL   512          // N_OUT * N_IN
#define PTS_PER_B 32768        // NWIN * NSEQ
#define N_PTS     131072       // B * PTS_PER_B

#define M_BLK   128
#define N_BLK   64
#define THREADS 256
#define KPAD    (K_DIM + 8)    // pad LDS rows to rotate banks

// scale = (T/2)^-0.5 with T=127 ; const0 = scale/sqrt(2) ; omega = 2*pi/127
#define F_SCALE  0.12549116f
#define F_CONST0 0.08873574f
#define F_OMEGA  0.04947390f

__global__ __launch_bounds__(THREADS)
void fourier_wmma_kernel(const float* __restrict__ x,
                         const float* __restrict__ coefs,
                         float* __restrict__ out)
{
    __shared__ _Float16 sW[M_BLK][KPAD];   // A operand slice (f16)
    __shared__ _Float16 sB[N_BLK][KPAD];   // basis (B operand, f16)
    __shared__ float    sBias[M_BLK];

    const int tid  = threadIdx.x;
    const int lane = tid & 31;
    const int wave = tid >> 5;
    const int mg   = blockIdx.y;           // 0..3   (row group of 128)
    const int P0   = blockIdx.x * N_BLK;   // point base (within one b)

    // ---- stage W rows (f32 -> f16): 2 threads per row, 64 halves each ----
    {
        const int row  = tid >> 1;                       // 0..127
        const int hsel = (tid & 1) * 64;                 // 0 or 64
        const float* src = coefs + (size_t)(mg * M_BLK + row) * N_BASIS + 1 + hsel;
        #pragma unroll 8
        for (int j = 0; j < 64; ++j)
            sW[row][hsel + j] = (_Float16)src[j];
    }
    if (tid < M_BLK)
        sBias[tid] = F_CONST0 * coefs[(size_t)(mg * M_BLK + tid) * N_BASIS];

    // ---- basis: 4 threads per point, 16 harmonics each via recurrence ----
    {
        const int p   = tid >> 2;                        // 0..63
        const int sub = tid & 3;                         // 0..3
        const float th = F_OMEGA * x[P0 + p];
        const float n0 = (float)(sub * 16 + 1);
        float sc = __sinf(n0 * th);
        float cs = __cosf(n0 * th);
        const float s1 = __sinf(th);
        const float c1 = __cosf(th);
        _Float16* dst = &sB[p][sub * 32];                // k = 2*(n-1), n = sub*16+1..
        #pragma unroll
        for (int j = 0; j < 16; ++j) {
            dst[2 * j]     = (_Float16)(sc * F_SCALE);   // sin(n*th)
            dst[2 * j + 1] = (_Float16)(cs * F_SCALE);   // cos(n*th)
            const float ns = sc * c1 + cs * s1;
            const float nc = cs * c1 - sc * s1;
            sc = ns; cs = nc;
        }
    }
    __syncthreads();

    // ---- WMMA: wave owns 16-row strip (mt = wave), 4 N-tiles of 16 pts ----
    v8f acc[4] = {v8f{}, v8f{}, v8f{}, v8f{}};
    const int l15  = lane & 15;
    const int hi   = (lane >= 16) ? 1 : 0;
    const int arow = wave * 16 + l15;                    // A: M = lane&15

    #pragma unroll
    for (int ks = 0; ks < 4; ++ks) {                     // K step of 32
        v16h a;
        {
            // A layout: lane<16 -> K = ks*32 + {0..7, 16..23}; lane>=16 -> +8
            const _Float16* wr = &sW[arow][ks * 32 + (hi ? 8 : 0)];
            #pragma unroll
            for (int h = 0; h < 8; ++h) a[h]     = wr[h];
            #pragma unroll
            for (int h = 0; h < 8; ++h) a[8 + h] = wr[16 + h];
        }
        #pragma unroll
        for (int nt = 0; nt < 4; ++nt) {
            // B layout: N = lane&15; lanes 0-15 K = ks*32 + {0..7, 16..23}, lanes 16-31 +8
            v16h bm;
            const _Float16* br = &sB[nt * 16 + l15][ks * 32 + (hi ? 8 : 0)];
            #pragma unroll
            for (int h = 0; h < 16; ++h) bm[h] = br[(h & 7) + ((h >> 3) << 4)];
            acc[nt] = __builtin_amdgcn_wmma_f32_16x16x32_f16(
                false, a, false, bm, (short)0, acc[nt], false, false);
        }
    }

    // ---- epilogue: bias add + coalesced f32 stores ----
    const int bidx = P0 / PTS_PER_B;
    const int pin0 = P0 % PTS_PER_B;
    #pragma unroll
    for (int nt = 0; nt < 4; ++nt) {
        const int pin = pin0 + nt * 16 + l15;            // D: N = lane&15
        #pragma unroll
        for (int r = 0; r < 8; ++r) {
            const int mloc  = wave * 16 + r + (hi ? 8 : 0);  // D: M = r + 8*hi
            const int mglob = mg * M_BLK + mloc;
            out[((size_t)bidx * M_TOTAL + mglob) * (size_t)PTS_PER_B + pin] =
                acc[nt][r] + sBias[mloc];
        }
    }
}

extern "C" void kernel_launch(void* const* d_in, const int* in_sizes, int n_in,
                              void* d_out, int out_size, void* d_ws, size_t ws_size,
                              hipStream_t stream) {
    (void)in_sizes; (void)n_in; (void)out_size; (void)d_ws; (void)ws_size;
    const float* x     = (const float*)d_in[0];
    const float* coefs = (const float*)d_in[1];
    float* out         = (float*)d_out;
    dim3 grid(N_PTS / N_BLK, M_TOTAL / M_BLK);           // (2048, 4)
    fourier_wmma_kernel<<<grid, THREADS, 0, stream>>>(x, coefs, out);
}
